// MultiHeadAttention_75110388072857
// MI455X (gfx1250) — matrix operand order repaired and checked by reading, not compiled, not run on hardware
//
#include <hip/hip_runtime.h>
#include <hip/hip_bf16.h>
#include <math.h>

typedef __bf16 bf16;
typedef __attribute__((ext_vector_type(16))) __bf16 v16bf;
typedef __attribute__((ext_vector_type(8)))  float  v8f;
typedef __attribute__((ext_vector_type(4)))  int    v4i;

union F8      { v8f v; float f[8]; };
union BF16x16 { v16bf v; unsigned u[8]; bf16 e[16]; };

#define WMMA_BF16(a, b, c) \
  __builtin_amdgcn_wmma_f32_16x16x32_bf16(false, (a), false, (b), (short)0, (c), false, false)

// ---------------- async global->LDS copies (CDNA5 ASYNCcnt path) ----------------
#if __has_builtin(__builtin_amdgcn_global_load_async_to_lds_b128) && \
    __has_builtin(__builtin_amdgcn_s_wait_asynccnt)
#define HAVE_ASYNC_LDS 1
#else
#define HAVE_ASYNC_LDS 0
#endif

static __device__ inline void async16(void* lds, const void* g) {
#if HAVE_ASYNC_LDS
  __builtin_amdgcn_global_load_async_to_lds_b128(
      (__attribute__((address_space(1))) v4i*)(void*)g,
      (__attribute__((address_space(3))) v4i*)lds, 0, 0);
#else
  *(uint4*)lds = *(const uint4*)g;
#endif
}
static __device__ inline void async_join() {
#if HAVE_ASYNC_LDS
  __builtin_amdgcn_s_wait_asynccnt(0);
#endif
}

// ---------------- fragment loaders (ISA 7.12.2 layouts, wave32) ----------------

// A fragment 16x32 bf16 from row-major [16 x >=32].
// lane<16: row=lane, K={0..7,16..23}; lane>=16: row=lane-16, K={8..15,24..31}.
static __device__ inline v16bf frag_a_lds(const bf16* s, int ld) {
  int lane = threadIdx.x & 31;
  const bf16* p = s + (size_t)(lane & 15) * ld;
  int kb = (lane >> 4) << 3;
  BF16x16 r;
#pragma unroll
  for (int i = 0; i < 8; ++i) {
    int k = ((i >> 2) << 4) + kb + ((i & 3) << 1);
    r.u[i] = *(const unsigned*)(p + k);  // K pair (k, k+1)
  }
  return r.v;
}

// B fragment 32x16 from K-contiguous (transposed) storage: element (k,col)=s[col*ld+k].
// lane<16: col=lane, K={0..7,16..23}; lane>=16: col=lane-16, K={8..15,24..31}. Pairs -> u32 reads.
static __device__ inline v16bf frag_bt_lds(const bf16* s, int ld) {
  int lane = threadIdx.x & 31;
  const bf16* p = s + (size_t)(lane & 15) * ld + ((lane >> 4) << 3);
  BF16x16 r;
#pragma unroll
  for (int i = 0; i < 8; ++i) r.u[i] = *(const unsigned*)(p + ((i >> 2) << 4) + ((i & 3) << 1));
  return r.v;
}

// ---------------- f32 [K][N] -> bf16 [N][K] tiled transpose-convert ----------------
__global__ __launch_bounds__(256) void mha_cvtT_kernel(const float* __restrict__ in,
                                                       bf16* __restrict__ out,
                                                       int K, int N) {
  __shared__ float sT[32][33];
  int k0 = blockIdx.y * 32, n0 = blockIdx.x * 32;
  int tid = threadIdx.x;
  int r = tid >> 5, c = tid & 31;
#pragma unroll
  for (int j = 0; j < 4; ++j)
    sT[r + j * 8][c] = in[(size_t)(k0 + r + j * 8) * N + n0 + c];
  __syncthreads();
#pragma unroll
  for (int j = 0; j < 4; ++j)
    out[(size_t)(n0 + r + j * 8) * K + k0 + c] = (bf16)sT[c][r + j * 8];
}

// ---------------- LayerNorm: one block per token, D=1024 ----------------
__global__ __launch_bounds__(256) void mha_ln_kernel(const float* __restrict__ x,
                                                     const float* __restrict__ w,
                                                     const float* __restrict__ bch,
                                                     bf16* __restrict__ xn) {
  __shared__ float red[256];
  int t = blockIdx.x, tid = threadIdx.x;
  const float4 xv = *(const float4*)(x + (size_t)t * 1024 + tid * 4);
  red[tid] = xv.x + xv.y + xv.z + xv.w;
  __syncthreads();
#pragma unroll
  for (int off = 128; off > 0; off >>= 1) {
    if (tid < off) red[tid] += red[tid + off];
    __syncthreads();
  }
  float mu = red[0] * (1.0f / 1024.0f);
  __syncthreads();
  float dx = xv.x - mu, dy = xv.y - mu, dz = xv.z - mu, dw = xv.w - mu;
  red[tid] = dx * dx + dy * dy + dz * dz + dw * dw;
  __syncthreads();
#pragma unroll
  for (int off = 128; off > 0; off >>= 1) {
    if (tid < off) red[tid] += red[tid + off];
    __syncthreads();
  }
  float rs = rsqrtf(red[0] * (1.0f / 1024.0f) + 1e-5f);
  const float4 wv = *(const float4*)(w + tid * 4);
  const float4 bv = *(const float4*)(bch + tid * 4);
  bf16* op = xn + (size_t)t * 1024 + tid * 4;
  op[0] = (bf16)(dx * rs * wv.x + bv.x);
  op[1] = (bf16)(dy * rs * wv.y + bv.y);
  op[2] = (bf16)(dz * rs * wv.z + bv.z);
  op[3] = (bf16)(dw * rs * wv.w + bv.w);
}

// ---------------- bf16 GEMM: C[M,N] = A[M,K] * Bt[N,K]^T + bias ----------------
// mode 0: f32 row-major output.  mode 1: scatter Q,K -> [B,H,S,DH], V -> [B,H,DH,S].
__global__ __launch_bounds__(256) void mha_gemm_kernel(
    const bf16* __restrict__ A, const bf16* __restrict__ Bt,
    const float* __restrict__ bias, float* __restrict__ Cf,
    bf16* __restrict__ Qo, bf16* __restrict__ Ko, bf16* __restrict__ Vo,
    int M, int N, int K, int mode) {
  __shared__ bf16 sA[128][48];   // 128x32, stride 48 (96B rows, 16B-aligned)
  __shared__ bf16 sBt[64][40];   // 64(n)x32(k), stride 40 (80B rows, 16B-aligned)

  int tid = threadIdx.x, lane = tid & 31, wid = tid >> 5;
  int cl = lane & 15, half = lane >> 4;
  int bm = blockIdx.y * 128, bn = blockIdx.x * 64;
  int wm = (wid & 3) << 5, wn = (wid >> 2) << 5;  // wave tile 32x32

  F8 acc[2][2];
#pragma unroll
  for (int mt = 0; mt < 2; ++mt)
#pragma unroll
    for (int nt = 0; nt < 2; ++nt) acc[mt][nt].v = 0;

  int ra = tid >> 1, ca = (tid & 1) << 4;  // A: 16 bf16 per thread
  int rb = tid >> 2, cb = (tid & 3) << 3;  // Bt: 8 bf16 per thread

  for (int k0 = 0; k0 < K; k0 += 32) {
    __syncthreads();
    const bf16* ga = A + (size_t)(bm + ra) * K + k0 + ca;
    async16(&sA[ra][ca], ga);
    async16(&sA[ra][ca + 8], ga + 8);
    async16(&sBt[rb][cb], Bt + (size_t)(bn + rb) * K + k0 + cb);
    async_join();
    __syncthreads();

    v16bf a0 = frag_a_lds(&sA[wm][0], 48);
    v16bf a1 = frag_a_lds(&sA[wm + 16][0], 48);
    v16bf b0 = frag_bt_lds(&sBt[wn][0], 40);
    v16bf b1 = frag_bt_lds(&sBt[wn + 16][0], 40);
    acc[0][0].v = WMMA_BF16(a0, b0, acc[0][0].v);
    acc[0][1].v = WMMA_BF16(a0, b1, acc[0][1].v);
    acc[1][0].v = WMMA_BF16(a1, b0, acc[1][0].v);
    acc[1][1].v = WMMA_BF16(a1, b1, acc[1][1].v);
  }

#pragma unroll
  for (int mt = 0; mt < 2; ++mt) {
    int row0 = bm + wm + mt * 16;
    int bb = row0 >> 11;                       // uniform in 16-row tile
    int ss0 = (row0 & 2047) + (half << 3);     // + r below
#pragma unroll
    for (int nt = 0; nt < 2; ++nt) {
      int c0 = bn + wn + nt * 16;              // 16-aligned, uniform per tile
      float bv = bias[c0 + cl];
      if (mode == 0) {
        float* cp = Cf + (size_t)(row0 + (half << 3)) * N + c0 + cl;
#pragma unroll
        for (int r = 0; r < 8; ++r) cp[(size_t)r * N] = acc[mt][nt].f[r] + bv;
      } else {
        int g = c0 >> 10;                      // uniform: 0=Q,1=K,2=V
        int rem = c0 & 1023;
        int hh = rem >> 6;                     // uniform head
        int dh = (rem & 63) + cl;
        if (g == 2) {
          // V transposed: [B,H,DH,S]; 8 consecutive s -> one b128 store
          bf16* vp = Vo + (((size_t)bb * 16 + hh) * 64 + dh) * 2048 + ss0;
          BF16x16 pk;
#pragma unroll
          for (int r = 0; r < 8; ++r) pk.e[r] = (bf16)(acc[mt][nt].f[r] + bv);
          *(uint4*)vp = *(uint4*)&pk.u[0];
        } else {
          bf16* qp = (g == 0 ? Qo : Ko) + (((size_t)bb * 16 + hh) * 2048 + ss0) * 64 + dh;
#pragma unroll
          for (int r = 0; r < 8; ++r) qp[(size_t)r * 64] = (bf16)(acc[mt][nt].f[r] + bv);
        }
      }
    }
  }
}

// ---------------- flash attention: block = (b,h, 128 q rows), 8 waves ----------------
__global__ __launch_bounds__(256) void mha_attn_kernel(
    const bf16* __restrict__ Q, const bf16* __restrict__ Kg,
    const bf16* __restrict__ Vt, const int* __restrict__ mask,
    bf16* __restrict__ ctx) {
  const int S = 2048;
  __shared__ bf16 sK[32][72];      // 32 keys x 64 dh
  __shared__ bf16 sVt[64][40];     // 64 dh x 32 keys (transposed V)
  __shared__ bf16 sP[8][16][40];   // per-wave probs 16x32

  int tid = threadIdx.x, lane = tid & 31, wid = tid >> 5;
  int cl = lane & 15, half = lane >> 4;
  int bh = blockIdx.y;
  int b = bh >> 4, h = bh & 15;
  int qbase = blockIdx.x * 128 + wid * 16;

  // Q fragments for this wave's 16 rows (two K=32 fragments over DH=64)
  int qrow = qbase + cl;
  const bf16* qp = Q + ((size_t)bh * S + qrow) * 64;
  int kb8 = half << 3;
  BF16x16 qa0, qa1;
#pragma unroll
  for (int i = 0; i < 8; ++i) {
    int k = ((i >> 2) << 4) + kb8 + ((i & 3) << 1);
    qa0.u[i] = *(const unsigned*)(qp + k);
    qa1.u[i] = *(const unsigned*)(qp + 32 + k);
  }

  F8 o[4];
#pragma unroll
  for (int d = 0; d < 4; ++d) o[d].v = 0;
  float m_r[8], l_r[8];
#pragma unroll
  for (int r = 0; r < 8; ++r) { m_r[r] = -1e30f; l_r[r] = 0.0f; }

  const float scale = 0.125f;  // 64^-0.5
  const int* mbase = mask + (size_t)b * S * S;

  int rk = tid >> 3, ck = (tid & 7) << 3;  // K tile: 32 x 64
  int rv = tid >> 2, cv = (tid & 3) << 3;  // Vt tile: 64 x 32

  for (int kt = 0; kt < S / 32; ++kt) {
    const bf16* kgp = Kg + ((size_t)bh * S + kt * 32 + rk) * 64 + ck;
    const bf16* vgp = Vt + ((size_t)bh * 64 + rv) * S + kt * 32 + cv;
    __syncthreads();
    async16(&sK[rk][ck], kgp);
    async16(&sVt[rv][cv], vgp);
    async_join();
    __syncthreads();
    if (kt + 1 < S / 32) {
      __builtin_prefetch(kgp + 32 * 64, 0, 1);
      __builtin_prefetch(vgp + 32, 0, 1);
    }

    // scores: Q (16x64) @ K^T (64x32) -> two 16x16 tiles
    F8 sc[2];
#pragma unroll
    for (int nt = 0; nt < 2; ++nt) {
      v8f c = 0;
      v16bf bt0 = frag_bt_lds(&sK[nt * 16][0], 72);
      v16bf bt1 = frag_bt_lds(&sK[nt * 16][32], 72);
      c = WMMA_BF16(qa0.v, bt0, c);
      c = WMMA_BF16(qa1.v, bt1, c);
      sc[nt].v = c;
    }
    // scale + mask
#pragma unroll
    for (int nt = 0; nt < 2; ++nt) {
      int colk = kt * 32 + nt * 16 + cl;
#pragma unroll
      for (int r = 0; r < 8; ++r) {
        int srow = qbase + (half << 3) + r;
        float sv = sc[nt].f[r] * scale;
        sv = mbase[(size_t)srow * S + colk] ? sv : -1e9f;
        sc[nt].f[r] = sv;
      }
    }
    // online softmax (row reductions across 16-lane halves)
#pragma unroll
    for (int r = 0; r < 8; ++r) {
      float mx = fmaxf(sc[0].f[r], sc[1].f[r]);
#pragma unroll
      for (int j = 1; j < 16; j <<= 1) mx = fmaxf(mx, __shfl_xor(mx, j, 32));
      float mn = fmaxf(m_r[r], mx);
      float corr = __expf(m_r[r] - mn);
      float p0 = __expf(sc[0].f[r] - mn);
      float p1 = __expf(sc[1].f[r] - mn);
      sc[0].f[r] = p0; sc[1].f[r] = p1;
      float ps = p0 + p1;
#pragma unroll
      for (int j = 1; j < 16; j <<= 1) ps += __shfl_xor(ps, j, 32);
      l_r[r] = l_r[r] * corr + ps;
      m_r[r] = mn;
#pragma unroll
      for (int d = 0; d < 4; ++d) o[d].f[r] *= corr;
    }
    // stage P as bf16 A-fragment via wave-private LDS (DS ops in-order per wave)
#pragma unroll
    for (int nt = 0; nt < 2; ++nt)
#pragma unroll
      for (int r = 0; r < 8; ++r)
        sP[wid][(half << 3) + r][nt * 16 + cl] = (bf16)sc[nt].f[r];

    v16bf pa = frag_a_lds(&sP[wid][0][0], 40);
#pragma unroll
    for (int d = 0; d < 4; ++d) {
      v16bf vb = frag_bt_lds(&sVt[d * 16][0], 40);
      o[d].v = WMMA_BF16(pa, vb, o[d].v);
    }
  }

  // epilogue: normalize and write ctx[B*S, H*DH] bf16
#pragma unroll
  for (int r = 0; r < 8; ++r) {
    float inv = 1.0f / l_r[r];
    int srow = qbase + (half << 3) + r;
    size_t tok = (size_t)b * S + srow;
#pragma unroll
    for (int d = 0; d < 4; ++d)
      ctx[tok * 1024 + h * 64 + d * 16 + cl] = (bf16)(o[d].f[r] * inv);
  }
}

// ---------------- launcher ----------------
extern "C" void kernel_launch(void* const* d_in, const int* in_sizes, int n_in,
                              void* d_out, int out_size, void* d_ws, size_t ws_size,
                              hipStream_t stream) {
  const float* x     = (const float*)d_in[0];
  const int*   mask  = (const int*)d_in[1];
  const float* ln_w  = (const float*)d_in[2];
  const float* ln_b  = (const float*)d_in[3];
  const float* w_qkv = (const float*)d_in[4];
  const float* b_qkv = (const float*)d_in[5];
  const float* w_out = (const float*)d_in[6];
  const float* b_out = (const float*)d_in[7];
  float* out = (float*)d_out;

  char* ws = (char*)d_ws;
  bf16* XN    = (bf16*)(ws);                            // 8 MiB  (4096x1024)
  bf16* WQKVt = (bf16*)(ws + (size_t)8  * 1024 * 1024); // 6 MiB  (3072x1024, transposed)
  bf16* WOUTt = (bf16*)(ws + (size_t)14 * 1024 * 1024); // 2 MiB  (1024x1024, transposed)
  bf16* Qb    = (bf16*)(ws + (size_t)16 * 1024 * 1024); // 8 MiB  [B,H,S,DH]
  bf16* Kb    = (bf16*)(ws + (size_t)24 * 1024 * 1024); // 8 MiB  [B,H,S,DH]
  bf16* Vb    = (bf16*)(ws + (size_t)32 * 1024 * 1024); // 8 MiB  [B,H,DH,S]
  bf16* CTX   = (bf16*)(ws + (size_t)40 * 1024 * 1024); // 8 MiB  [B*S, H*DH]

  mha_cvtT_kernel<<<dim3(96, 32), 256, 0, stream>>>(w_qkv, WQKVt, 1024, 3072);
  mha_cvtT_kernel<<<dim3(32, 32), 256, 0, stream>>>(w_out, WOUTt, 1024, 1024);
  mha_ln_kernel<<<4096, 256, 0, stream>>>(x, ln_w, ln_b, XN);
  mha_gemm_kernel<<<dim3(48, 32), 256, 0, stream>>>(XN, WQKVt, b_qkv, nullptr,
                                                    Qb, Kb, Vb, 4096, 3072, 1024, 1);
  mha_attn_kernel<<<dim3(16, 32), 256, 0, stream>>>(Qb, Kb, Vb, mask, CTX);
  mha_gemm_kernel<<<dim3(16, 32), 256, 0, stream>>>(CTX, WOUTt, b_out, out,
                                                    nullptr, nullptr, nullptr, 4096, 1024, 1024, 0);
}
